// LinearSim_44985487458619
// MI455X (gfx1250) — matrix operand order repaired and checked by reading, not compiled, not run on hardware
//
#include <hip/hip_runtime.h>
#include <math.h>

typedef float    v8f  __attribute__((ext_vector_type(8)));
typedef _Float16 v4h  __attribute__((ext_vector_type(4)));
typedef _Float16 v8h  __attribute__((ext_vector_type(8)));
typedef _Float16 v16h __attribute__((ext_vector_type(16)));

#define M_TOK  65536
#define N_FEAT 512       // in_features  (GEMM K)
#define K_OUT  512       // out_features (GEMM N)
#define ROWS   32        // token rows per block
#define XS     520       // LDS row stride in halves (512 + 8 pad -> conflict-free)

// --- monotonic float <-> unsigned key for atomic max over signed floats ---
__device__ __forceinline__ unsigned f32_key(float f) {
    unsigned u = __float_as_uint(f);
    return (u & 0x80000000u) ? ~u : (u | 0x80000000u);
}
__device__ __forceinline__ float key_f32(unsigned k) {
    return (k & 0x80000000u) ? __uint_as_float(k ^ 0x80000000u)
                             : __uint_as_float(~k);
}

__global__ void ls_init_keys(unsigned* keys) {
    if (threadIdx.x == 0) { keys[0] = 0u; keys[1] = 0u; }
}

__global__ void ls_reduce_max(const float* __restrict__ in, int n4,
                              unsigned* __restrict__ out_key, int use_abs) {
    int idx = blockIdx.x * blockDim.x + threadIdx.x;
    int stride = gridDim.x * blockDim.x;
    const float4* in4 = (const float4*)in;
    float m = -INFINITY;
    for (int i = idx; i < n4; i += stride) {
        float4 v = in4[i];
        float a = v.x, b = v.y, c = v.z, d = v.w;
        if (use_abs) { a = fabsf(a); b = fabsf(b); c = fabsf(c); d = fabsf(d); }
        m = fmaxf(m, fmaxf(fmaxf(a, b), fmaxf(c, d)));
    }
    #pragma unroll
    for (int off = 16; off > 0; off >>= 1)
        m = fmaxf(m, __shfl_xor(m, off, 32));
    if ((threadIdx.x & 31) == 0)
        atomicMax(out_key, f32_key(m));
}

// Normalize W and emit compensated-f16 pair (wh, wl) in B-fragment-swizzled
// layout: block (kt, n) -> 16 contiguous halves = W_norm[n][kt*16 .. kt*16+15].
// One thread per (kt, n) block: 32 * 512 = 16384 threads.
__global__ void ls_prep_w(const float* __restrict__ W,
                          const unsigned* __restrict__ keys,
                          _Float16* __restrict__ Wh, _Float16* __restrict__ Wl) {
    const float inv_nw = 1.0f / key_f32(keys[1]);
    int o = blockIdx.x * blockDim.x + threadIdx.x;   // 0 .. 16383
    if (o >= 32 * K_OUT) return;
    int n  = o & (K_OUT - 1);
    int kt = o >> 9;
    const float* src = W + (size_t)n * N_FEAT + kt * 16;
    _Float16* dh = Wh + ((size_t)o << 4);
    _Float16* dl = Wl + ((size_t)o << 4);
    #pragma unroll
    for (int j = 0; j < 16; ++j) {
        float wn = src[j] * inv_nw;
        _Float16 h = (_Float16)wn;
        dh[j] = h;
        dl[j] = (_Float16)(wn - (float)h);
    }
}

// Block = 32 token rows x all 512 out-features. 8 waves; wave w owns a
// 32x64 slab: 2 row-tiles x 4 col-tiles, one f32 accumulator per 16x16 tile.
// Per 32-deep k-step: 3 chained v_wmma_f32_16x16x32_f16 per tile
// (xh*wh + xh*wl + xl*wh).
__global__ __launch_bounds__(256)
void ls_gemm_f16c(const float* __restrict__ X,
                  const _Float16* __restrict__ Wh,
                  const _Float16* __restrict__ Wl,
                  const float* __restrict__ Bias,
                  float* __restrict__ out,
                  const unsigned* __restrict__ keys) {
    __shared__ _Float16 xh[ROWS * XS];
    __shared__ _Float16 xl[ROWS * XS];

    const float n_x = key_f32(keys[0]);
    const float n_w = key_f32(keys[1]);
    const float inv_nx = 1.0f / n_x;

    const int tid  = threadIdx.x;
    const int row0 = blockIdx.x * ROWS;

    // ---- stage X tile (32 x 512) as compensated f16 hi/lo into LDS ----
    {
        const float4* X4 = (const float4*)(X + (size_t)row0 * N_FEAT);
        #pragma unroll
        for (int i = 0; i < 16; ++i) {
            int id = i * 256 + tid;          // 0 .. 4095 float4 chunks
            int r  = id >> 7;                // 128 float4 per row
            int c  = (id & 127) << 2;
            float4 v = X4[id];
            float f0 = v.x * inv_nx, f1 = v.y * inv_nx;
            float f2 = v.z * inv_nx, f3 = v.w * inv_nx;
            _Float16 h0 = (_Float16)f0, h1 = (_Float16)f1;
            _Float16 h2 = (_Float16)f2, h3 = (_Float16)f3;
            v4h hv = {h0, h1, h2, h3};
            v4h lv = {(_Float16)(f0 - (float)h0), (_Float16)(f1 - (float)h1),
                      (_Float16)(f2 - (float)h2), (_Float16)(f3 - (float)h3)};
            *(v4h*)&xh[r * XS + c] = hv;
            *(v4h*)&xl[r * XS + c] = lv;
        }
    }
    __syncthreads();

    const int wave = tid >> 5;
    const int lane = tid & 31;
    const int hi   = lane >> 4;     // half-wave select
    const int l15  = lane & 15;
    const int colBase = wave * 64;

    v8f acc[2][4] = {};

    for (int kb = 0; kb < N_FEAT; kb += 32) {
        // A fragments (contiguous k order, same as B): lanes 0-15 -> K {kb..kb+15},
        // lanes 16-31 -> K {kb+16..kb+31}
        v16h ah[2], al[2];
        #pragma unroll
        for (int mt = 0; mt < 2; ++mt) {
            const _Float16* bh_ = &xh[(mt * 16 + l15) * XS + kb + hi * 16];
            v8h h0 = *(const v8h*)bh_;
            v8h h1 = *(const v8h*)(bh_ + 8);
            ah[mt] = __builtin_shufflevector(h0, h1, 0, 1, 2, 3, 4, 5, 6, 7,
                                             8, 9, 10, 11, 12, 13, 14, 15);
            const _Float16* bl_ = &xl[(mt * 16 + l15) * XS + kb + hi * 16];
            v8h l0 = *(const v8h*)bl_;
            v8h l1 = *(const v8h*)(bl_ + 8);
            al[mt] = __builtin_shufflevector(l0, l1, 0, 1, 2, 3, 4, 5, 6, 7,
                                             8, 9, 10, 11, 12, 13, 14, 15);
        }

        const int kt = (kb >> 4) + hi;  // lanes 0-15: K 0-15; lanes 16-31: K 16-31
        #pragma unroll
        for (int nt = 0; nt < 4; ++nt) {
            const int n = colBase + nt * 16 + l15;
            const size_t off = ((size_t)(kt * K_OUT + n)) << 4;
            v16h bh = *(const v16h*)(Wh + off);
            v16h bl = *(const v16h*)(Wl + off);
            #pragma unroll
            for (int mt = 0; mt < 2; ++mt) {
                acc[mt][nt] = __builtin_amdgcn_wmma_f32_16x16x32_f16(
                    false, ah[mt], false, bh, (short)0, acc[mt][nt], false, false);
                acc[mt][nt] = __builtin_amdgcn_wmma_f32_16x16x32_f16(
                    false, ah[mt], false, bl, (short)0, acc[mt][nt], false, false);
                acc[mt][nt] = __builtin_amdgcn_wmma_f32_16x16x32_f16(
                    false, al[mt], false, bh, (short)0, acc[mt][nt], false, false);
            }
        }
    }

    // ---- epilogue: out = n_x*n_w*acc + bias ----
    const float scale = n_x * n_w;
    #pragma unroll
    for (int mt = 0; mt < 2; ++mt) {
        #pragma unroll
        for (int nt = 0; nt < 4; ++nt) {
            const int n = colBase + nt * 16 + l15;
            const float b = Bias[n];
            #pragma unroll
            for (int r = 0; r < 8; ++r) {
                const int mm = row0 + mt * 16 + r + (hi << 3);
                out[(size_t)mm * K_OUT + n] = scale * acc[mt][nt][r] + b;
            }
        }
    }
}

extern "C" void kernel_launch(void* const* d_in, const int* in_sizes, int n_in,
                              void* d_out, int out_size, void* d_ws, size_t ws_size,
                              hipStream_t stream) {
    const float* X    = (const float*)d_in[0];
    const float* W    = (const float*)d_in[1];
    const float* Bias = (const float*)d_in[2];
    float* out        = (float*)d_out;

    unsigned* keys = (unsigned*)d_ws;
    _Float16* Wh   = (_Float16*)((char*)d_ws + 64);
    _Float16* Wl   = Wh + (size_t)K_OUT * N_FEAT;      // ~1.05 MB of d_ws total

    ls_init_keys<<<1, 32, 0, stream>>>(keys);
    ls_reduce_max<<<2048, 256, 0, stream>>>(X, (M_TOK * N_FEAT) / 4, keys + 0, 0);
    ls_reduce_max<<<64,   256, 0, stream>>>(W, (K_OUT * N_FEAT) / 4, keys + 1, 1);
    ls_prep_w<<<(32 * K_OUT) / 256, 256, 0, stream>>>(W, keys, Wh, Wl);
    ls_gemm_f16c<<<M_TOK / ROWS, 256, 0, stream>>>(X, Wh, Wl, Bias, out, keys);
}
